// LDM_60748017435171
// MI455X (gfx1250) — matrix operand order repaired and checked by reading, not compiled, not run on hardware
//
#include <hip/hip_runtime.h>

#define NEDGES 1000000
#define NB     12288
#define ND     16
#define NTILE  (NB / 16)   // 768 row/col tiles

typedef __attribute__((ext_vector_type(16))) _Float16 v16h;
typedef __attribute__((ext_vector_type(8)))  float    v8f;

union HV16 { v16h v; _Float16 h[16]; uint4 q[2]; };
union FV8  { v8f  v; float f[8]; };

__device__ __forceinline__ float warp_reduce(float x) {
#pragma unroll
  for (int o = 16; o > 0; o >>= 1) x += __shfl_xor(x, o, 32);
  return x;
}

__device__ __forceinline__ void block_reduce_atomic(float x, float* dst) {
  __shared__ float s[8];
  const int lane = threadIdx.x & 31;
  const int wave = threadIdx.x >> 5;
  x = warp_reduce(x);
  if (lane == 0) s[wave] = x;
  __syncthreads();
  if (wave == 0) {
    float y = (lane < (int)(blockDim.x >> 5)) ? s[lane] : 0.f;
    y = warp_reduce(y);
    if (lane == 0)
      __hip_atomic_fetch_add(dst, y, __ATOMIC_RELAXED, __HIP_MEMORY_SCOPE_AGENT);
  }
}

// ---------------------------------------------------------------------------
// Prep: gather sampled nodes -> f16 tile buffer + packed (||z||^2, beta) pairs.
// Thread 0 zeroes the accumulators (stream-ordered before their first use).
// ---------------------------------------------------------------------------
__global__ void ldm_prep(const int* __restrict__ nodes,
                         const float* __restrict__ beta,
                         const float* __restrict__ z,
                         float* __restrict__ acc,
                         float2* __restrict__ sb,
                         _Float16* __restrict__ zzc) {
  const int b = blockIdx.x * blockDim.x + threadIdx.x;
  if (b == 0) { acc[0] = 0.f; acc[1] = 0.f; }
  if (b >= NB) return;
  const int idx = nodes[b];
  const float4* zr = reinterpret_cast<const float4*>(z + (size_t)idx * ND);
  _Float16* out = zzc + (size_t)b * ND;
  float s = 0.f;
#pragma unroll
  for (int q = 0; q < 4; ++q) {
    const float4 t = zr[q];
    s = fmaf(t.x, t.x, s); s = fmaf(t.y, t.y, s);
    s = fmaf(t.z, t.z, s); s = fmaf(t.w, t.w, s);
    out[4 * q + 0] = (_Float16)t.x;
    out[4 * q + 1] = (_Float16)t.y;
    out[4 * q + 2] = (_Float16)t.z;
    out[4 * q + 3] = (_Float16)t.w;
  }
  sb[b] = make_float2(s, beta[idx]);
}

// ---------------------------------------------------------------------------
// Link term: sum_e w_e * (beta_i + beta_j - || z_i - z_j + 1e-6 ||)
// ---------------------------------------------------------------------------
__global__ void __launch_bounds__(256)
ldm_link(const int* __restrict__ eu, const int* __restrict__ ev,
         const float* __restrict__ w, const float* __restrict__ beta,
         const float* __restrict__ z, float* __restrict__ acc) {
  float local = 0.f;
  for (int e = blockIdx.x * blockDim.x + threadIdx.x; e < NEDGES;
       e += gridDim.x * blockDim.x) {
    const int i = eu[e], j = ev[e];
    const float bp = beta[i] + beta[j];
    const float4* zi = reinterpret_cast<const float4*>(z + (size_t)i * ND);
    const float4* zj = reinterpret_cast<const float4*>(z + (size_t)j * ND);
    float s = 0.f;
#pragma unroll
    for (int q = 0; q < 4; ++q) {
      const float4 a = zi[q];
      const float4 b = zj[q];
      const float dx = a.x - b.x + 1e-6f, dy = a.y - b.y + 1e-6f;
      const float dz = a.z - b.z + 1e-6f, dw = a.w - b.w + 1e-6f;
      s = fmaf(dx, dx, s); s = fmaf(dy, dy, s);
      s = fmaf(dz, dz, s); s = fmaf(dw, dw, s);
    }
    local += w[e] * (bp - __builtin_amdgcn_sqrtf(s));
  }
  block_reduce_atomic(local, acc + 0);
}

// ---------------------------------------------------------------------------
// Non-link term: each wave owns a 16x64 strip = 4 consecutive 16x16 pair
// tiles -> 4x v_wmma_f32_16x16x32_f16 reusing one A fragment. Branchless
// triangle mask via select; packed (sq,beta) pairs; raw v_sqrt/v_exp epilogue.
// grid = (NTILE/32, NTILE), block = 256 (8 waves).
// ---------------------------------------------------------------------------
__global__ void __launch_bounds__(256)
ldm_nonlink(const float2* __restrict__ sb, const _Float16* __restrict__ zzc,
            float* __restrict__ acc) {
  const int lane = threadIdx.x & 31;
  const int wave = threadIdx.x >> 5;
  const int ti = blockIdx.y;
  const int tjBase = (blockIdx.x * 8 + wave) * 4;
  float local = 0.f;

  if (tjBase + 3 >= ti) {  // wave-uniform: at least one sub-tile on/above diag
    const int i0 = ti * 16;

    // A fragment (16x32 f16, K=16..31 zero pad): lanes 0-15 carry M=lane,
    // K=0..7 in halves 0..7; lanes 16-31 carry K=8..15.
    HV16 a;
    a.q[0] = *reinterpret_cast<const uint4*>(
        zzc + ((size_t)(i0 + (lane & 15)) * ND + 8 * (lane >> 4)));
    a.q[1] = make_uint4(0u, 0u, 0u, 0u);

    // Row (||z||^2, beta) pairs for this half-wave's 8 output rows.
    const int mbase = i0 + 8 * (lane >> 4);
    const float4* rp = reinterpret_cast<const float4*>(sb + mbase);
    float rs[8], rb[8];
#pragma unroll
    for (int p = 0; p < 4; ++p) {
      const float4 rr = rp[p];
      rs[2 * p + 0] = rr.x; rb[2 * p + 0] = rr.y;
      rs[2 * p + 1] = rr.z; rb[2 * p + 1] = rr.w;
    }

#pragma unroll
    for (int t = 0; t < 4; ++t) {
      const int tj = tjBase + t;
      if (tj < ti) continue;  // wave-uniform skip (only near the diagonal)
      const int j0 = tj * 16;

      // B fragment (32x16 f16): lanes 0-15 carry column N=lane, K=0..7 in
      // halves 0..7; lanes 16-31 carry K=8..15; halves 8..15 are the zero pad.
      HV16 b;
      b.q[0] = *reinterpret_cast<const uint4*>(
          zzc + ((size_t)(j0 + (lane & 15)) * ND + 8 * (lane >> 4)));
      b.q[1] = make_uint4(0u, 0u, 0u, 0u);

      FV8 c;
#pragma unroll
      for (int k = 0; k < 8; ++k) c.f[k] = 0.f;
      c.v = __builtin_amdgcn_wmma_f32_16x16x32_f16(
          false, a.v, false, b.v, (short)0, c.v, false, false);

      // C/D layout: lane l holds N = l&15, M = v + 8*(l>>4) in VGPR v.
      const int gj = j0 + (lane & 15);
      const float2 cb = sb[gj];
#pragma unroll
      for (int v = 0; v < 8; ++v) {
        float d2 = rs[v] + cb.x - 2.0f * c.f[v];
        d2 = fmaxf(d2, 1e-12f);  // == sqrt(max(max(d2,0),1e-12)) in reference
        const float term = __expf(rb[v] + cb.y - __builtin_amdgcn_sqrtf(d2));
        local += (gj > mbase + v) ? term : 0.f;  // strict upper triangle
      }
    }
  }
  block_reduce_atomic(local, acc + 1);
}

// ---------------------------------------------------------------------------
__global__ void ldm_final(const float* __restrict__ acc, float* __restrict__ out) {
  if (threadIdx.x == 0 && blockIdx.x == 0) out[0] = acc[1] - acc[0];
}

extern "C" void kernel_launch(void* const* d_in, const int* in_sizes, int n_in,
                              void* d_out, int out_size, void* d_ws, size_t ws_size,
                              hipStream_t stream) {
  const int*   edges   = (const int*)d_in[0];   // [2, E] flat
  const float* weights = (const float*)d_in[1]; // [E]
  const int*   nodes   = (const int*)d_in[2];   // [B]
  const float* beta    = (const float*)d_in[3]; // [N]
  const float* z       = (const float*)d_in[4]; // [N, 16]
  float* out = (float*)d_out;

  // Workspace layout (all offsets 16B-aligned):
  float*    acc = (float*)d_ws;           // [0]=link, [1]=non_link
  float2*   sb  = (float2*)(acc + 16);    // (||z||^2, beta) per sampled node
  _Float16* zzc = (_Float16*)(sb + NB);   // z[nodes] in f16, B*16 halves

  ldm_prep<<<(NB + 255) / 256, 256, 0, stream>>>(nodes, beta, z, acc, sb, zzc);
  ldm_link<<<1024, 256, 0, stream>>>(edges, edges + NEDGES, weights, beta, z, acc);
  dim3 grid(NTILE / 32, NTILE);
  ldm_nonlink<<<grid, 256, 0, stream>>>(sb, zzc, acc);
  ldm_final<<<1, 1, 0, stream>>>(acc, out);
}
